// VectorQuantizer_67087389163578
// MI455X (gfx1250) — matrix operand order repaired and checked by reading, not compiled, not run on hardware
//
#include <hip/hip_runtime.h>
#include <hip/hip_bf16.h>
#include <stdint.h>

typedef __attribute__((ext_vector_type(16))) __bf16 v16bf;
typedef __attribute__((ext_vector_type(8)))  __bf16 v8bf;
typedef __attribute__((ext_vector_type(8)))  float  v8f;

#define KCODES   1024
#define DDIM     256
#define NTOT     131072            // 32 * 4096
#define MTILE    128
#define ZSTRIDE  264               // 256 halves + 8-half pad (16B bank skew)
#define ESTRIDE  264
#define ZQ_ELEMS 33554432ull       // NTOT * DDIM
#define IDX_BASE 33554432ull
#define SCL_BASE 33685504ull       // ZQ + NTOT

// dynamic LDS partition (bytes), all 16B aligned
#define OFF_ZT   0                 // 128*264*2 = 67584
#define OFF_ET0  67584             // 16*264*2  = 8448
#define OFF_ET1  76032             // 16*264*2  = 8448
#define OFF_E2   84480             // 1024*4    = 4096
#define OFF_CS   88576             // 128*16*4  = 8192
#define OFF_CC   96768             // 128*16*4  = 8192
#define OFF_IDX  104960            // 128*4     = 512
#define SMEM_SZ  105472

// CDNA5 async DMA: global -> LDS, tracked by ASYNCcnt (ISA 08_async_tensor §4).
// GV mode: VDST = LDS byte offset (u32 VGPR), ADDR = 64-bit global addr (VGPR pair).
__device__ __forceinline__ void async_g2l_b128(uint32_t lds_byte, const void* gsrc) {
  asm volatile("global_load_async_to_lds_b128 %0, %1, off"
               :: "v"(lds_byte), "v"((uint64_t)(uintptr_t)gsrc) : "memory");
}
__device__ __forceinline__ void wait_asynccnt0() {
  asm volatile("s_wait_asynccnt 0x0" ::: "memory");
}

// ---------------- prep: bf16 embedding copy + ||e||^2, zero loss acc ----------
__global__ void vq_prep(const float* __restrict__ emb, __bf16* __restrict__ ebf,
                        float* __restrict__ e2, double* __restrict__ acc) {
  const int c = blockIdx.x, d = threadIdx.x;
  const float v = emb[(size_t)c * DDIM + d];
  ebf[(size_t)c * DDIM + d] = (__bf16)v;
  __shared__ float red[256];
  red[d] = v * v;
  __syncthreads();
  for (int s = 128; s > 0; s >>= 1) { if (d < s) red[d] += red[d + s]; __syncthreads(); }
  if (d == 0) e2[c] = red[0];
  if (c == 0 && d == 0) *acc = 0.0;
}

// ---------------- main: WMMA scores -> argmin -> gather -> loss ---------------
__global__ void __launch_bounds__(256) vq_main(
    const float* __restrict__ z_e, const float* __restrict__ emb,
    const __bf16* __restrict__ ebf, const float* __restrict__ e2g,
    float* __restrict__ out, double* __restrict__ lossAcc) {
  extern __shared__ char smem[];
  __bf16* zt   = (__bf16*)(smem + OFF_ZT);
  float*  e2s  = (float*) (smem + OFF_E2);
  float*  cs   = (float*) (smem + OFF_CS);
  int*    cc   = (int*)   (smem + OFF_CC);
  int*    idxs = (int*)   (smem + OFF_IDX);
  const uint32_t ldsbase = (uint32_t)(uintptr_t)smem;  // LDS aperture: low 32 bits = offset

  const int tid = threadIdx.x;
  const int n0  = blockIdx.x * MTILE;
  const int b   = n0 >> 12;
  const int l0  = n0 & 4095;
  const size_t zbase0 = (size_t)b * (DDIM * 4096) + l0;

  for (int i = tid; i < KCODES; i += 256) e2s[i] = e2g[i];

  // stage z tile as bf16 (coalesced over rows)
  {
    const int row = tid & 127;
    const int dof = tid >> 7;
    #pragma unroll 4
    for (int j = 0; j < 128; ++j) {
      const int d = dof + 2 * j;
      zt[row * ZSTRIDE + d] = (__bf16)z_e[zbase0 + (size_t)d * 4096 + row];
    }
  }

  // staging geometry: thread -> (row er, 16-half chunk ed), 32B per thread per tile
  const int er = tid >> 4;
  const int ed = (tid & 15) * 16;
  const uint32_t dstE0 = ldsbase + OFF_ET0 + (uint32_t)(er * ESTRIDE + ed) * 2;
  const uint32_t dstE1 = ldsbase + OFF_ET1 + (uint32_t)(er * ESTRIDE + ed) * 2;

  // prologue: async-stage e-tile 0 into buffer 0
  {
    const __bf16* src = ebf + (size_t)er * DDIM + ed;
    async_g2l_b128(dstE0,      src);
    async_g2l_b128(dstE0 + 16, src + 8);
    wait_asynccnt0();
  }
  __syncthreads();   // covers zt staging + e-tile 0

  const int lane = tid & 31, wv = tid >> 5;
  const int col  = lane & 15, side = lane >> 4;

  // resident A fragments: wave wv owns rows 16*wv .. 16*wv+15
  v16bf afrag[8];
  {
    const int abase = (wv * 16 + col) * ZSTRIDE + side * 8;
    #pragma unroll
    for (int k = 0; k < 8; ++k) {
      const v8bf lo = *(const v8bf*)&zt[abase + 32 * k];
      const v8bf hi = *(const v8bf*)&zt[abase + 32 * k + 16];
      afrag[k] = __builtin_shufflevector(lo, hi, 0,1,2,3,4,5,6,7,8,9,10,11,12,13,14,15);
    }
  }

  float bestS[8]; int bestC[8];
  #pragma unroll
  for (int i = 0; i < 8; ++i) { bestS[i] = -3.4e38f; bestC[i] = 0; }

  for (int t = 0; t < 64; ++t) {
    const int cur = t & 1;

    // issue async staging of tile t+1 into the other buffer (overlaps with WMMAs)
    if (t < 63) {
      const __bf16* src = ebf + (size_t)((t + 1) * 16 + er) * DDIM + ed;
      const uint32_t dst = cur ? dstE0 : dstE1;
      async_g2l_b128(dst,      src);
      async_g2l_b128(dst + 16, src + 8);
    }

    const __bf16* etb = (const __bf16*)(smem + (cur ? OFF_ET1 : OFF_ET0));
    const int c0 = t * 16;

    // preload all 8 B fragments (16 outstanding ds_load_b128)
    v16bf bfr[8];
    const int bb = col * ESTRIDE + side * 8;
    #pragma unroll
    for (int k = 0; k < 8; ++k) {
      const v8bf blo = *(const v8bf*)&etb[bb + 32 * k];
      const v8bf bhi = *(const v8bf*)&etb[bb + 32 * k + 16];
      bfr[k] = __builtin_shufflevector(blo, bhi, 0,1,2,3,4,5,6,7,8,9,10,11,12,13,14,15);
    }
    // fence: keep loads batched ahead of the WMMA chain (one s_wait_dscnt total)
    __builtin_amdgcn_sched_barrier(0);

    // accumulator starts at 0 (SRC2 inline 0); bias applied at compare time
    v8f acc = {};
    #pragma unroll
    for (int k = 0; k < 8; ++k)
      acc = __builtin_amdgcn_wmma_f32_16x16x32_bf16(false, afrag[k], false, bfr[k],
                                                    (short)0, acc, false, false);

    const float bias = -0.5f * e2s[c0 + col];   // off the WMMA critical path
    const int code = c0 + col;
    #pragma unroll
    for (int i = 0; i < 8; ++i) {
      const float s = acc[i] + bias;
      if (s > bestS[i]) { bestS[i] = s; bestC[i] = code; }
    }

    wait_asynccnt0();   // my async writes to the other buffer are in LDS
    __syncthreads();    // everyone done computing buf[t&1] + staging buf[(t+1)&1]
  }

  // per-lane candidates -> LDS
  #pragma unroll
  for (int i = 0; i < 8; ++i) {
    const int m = wv * 16 + side * 8 + i;
    cs[m * 16 + col] = bestS[i];
    cc[m * 16 + col] = bestC[i];
  }
  __syncthreads();

  // 16-way reduce per row; write index output (argmin, ties -> lowest code)
  if (tid < 128) {
    float bs = cs[tid * 16]; int bc = cc[tid * 16];
    #pragma unroll
    for (int j = 1; j < 16; ++j) {
      const float s = cs[tid * 16 + j]; const int c = cc[tid * 16 + j];
      if (s > bs || (s == bs && c < bc)) { bs = s; bc = c; }
    }
    idxs[tid] = bc;
    out[IDX_BASE + (size_t)(n0 + tid)] = (float)bc;
  }
  __syncthreads();

  // gather z_q (full f32) + squared-error accumulation
  float lsum = 0.f;
  {
    const int row = tid & 127;
    const int dof = tid >> 7;
    const float* erow = emb + (size_t)idxs[row] * DDIM;
    #pragma unroll 4
    for (int j = 0; j < 128; ++j) {
      const int d = dof + 2 * j;
      const size_t off = zbase0 + (size_t)d * 4096 + row;
      const float q = erow[d];
      const float z = z_e[off];
      out[off] = q;
      const float df = z - q;
      lsum += df * df;
    }
  }
  float* red = cs;  // reuse
  red[tid] = lsum;
  __syncthreads();
  for (int s = 128; s > 0; s >>= 1) { if (tid < s) red[tid] += red[tid + s]; __syncthreads(); }
  if (tid == 0) atomicAdd(lossAcc, (double)red[0]);
}

// ---------------- finalize scalars -------------------------------------------
__global__ void vq_final(const double* __restrict__ acc, float* __restrict__ out) {
  if (threadIdx.x == 0 && blockIdx.x == 0) {
    const float mse = (float)(*acc / (double)ZQ_ELEMS);
    out[SCL_BASE + 0] = mse + 0.25f * mse;  // vq_loss
    out[SCL_BASE + 1] = mse;                // codebook_loss
    out[SCL_BASE + 2] = 0.25f * mse;        // beta * commitment
  }
}

extern "C" void kernel_launch(void* const* d_in, const int* in_sizes, int n_in,
                              void* d_out, int out_size, void* d_ws, size_t ws_size,
                              hipStream_t stream) {
  const float* z_e = (const float*)d_in[0];
  const float* emb = (const float*)d_in[1];
  float* out = (float*)d_out;
  char* ws = (char*)d_ws;
  double* acc = (double*)ws;                  // 8 B
  float*  e2  = (float*)(ws + 64);            // 4 KB
  __bf16* ebf = (__bf16*)(ws + 64 + 4224);    // 512 KB, 16B aligned

  vq_prep <<<KCODES, 256, 0, stream>>>(emb, ebf, e2, acc);
  vq_main <<<NTOT / MTILE, 256, SMEM_SZ, stream>>>(z_e, emb, ebf, e2, out, acc);
  vq_final<<<1, 64, 0, stream>>>(acc, out);
}
